// BlockSlidingWindowAttention_89223650607231
// MI455X (gfx1250) — matrix operand order repaired and checked by reading, not compiled, not run on hardware
//
#include <hip/hip_runtime.h>
#include <hip/hip_bf16.h>

// ---------------------------------------------------------------------------
// Block sliding-window attention for MI455X (gfx1250).
// wave32 + v_wmma_f32_16x16x32_bf16 + Tensor Data Mover (LDS double buffer).
// ---------------------------------------------------------------------------

typedef __bf16 bf16;
typedef __attribute__((ext_vector_type(16))) __bf16 v16bf;
typedef __attribute__((ext_vector_type(8)))  __bf16 v8bf;
typedef __attribute__((ext_vector_type(4)))  __bf16 v4bf;
typedef __attribute__((ext_vector_type(8)))  float  v8f;
typedef __attribute__((ext_vector_type(4)))  float  v4f;
typedef __attribute__((ext_vector_type(4)))  unsigned int v4u;
typedef __attribute__((ext_vector_type(8)))  int    v8i;
typedef __attribute__((ext_vector_type(4)))  int    v4i;

#define DIMS     2048
#define HEADS    16
#define HEAD_DIM 128
#define WIN      256
#define SEQ      8192
#define BATCH    2
#define NTOK     (BATCH * SEQ)   // 16384 rows
#define NCHUNK   (SEQ / WIN)     // 32

__device__ __forceinline__ v16bf join8(v8bf lo, v8bf hi) {
  v16bf r;
#pragma unroll
  for (int i = 0; i < 8; ++i) { r[i] = lo[i]; r[i + 8] = hi[i]; }
  return r;
}

__device__ __forceinline__ v8f wmma_bf16(v16bf a, v16bf b, v8f c) {
  // D(16x16,f32) = A(16x32,bf16) * B(32x16,bf16) + C
  return __builtin_amdgcn_wmma_f32_16x16x32_bf16(
      /*neg_a=*/false, a, /*neg_b=*/false, b,
      /*c_mod=*/(short)0, c, /*reuse_a=*/false, /*reuse_b=*/false);
}

// LDS aperture rule: low 32 bits of a generic shared-space address are the
// LDS byte address (ISA 10.2).
__device__ __forceinline__ unsigned lds_addr32(const void* p) {
  return (unsigned)(uintptr_t)p;
}

// ---------------------------------------------------------------------------
// Tensor Data Mover: 2D bf16 tile load, Global -> LDS.
// D# per CDNA5 ISA 8.3/8.4: group0 {count=1, lds_addr, global_addr, type=2},
// group1 {data_size=2B, tensor_dim0/1, tile_dim0/1, tensor_dim0_stride}.
// tileD0 = contiguous (fastest) extent, tileD1 = rows; strideD0 in elements.
// This toolchain's builtin takes 6 args (extra int32x8 group before cpol).
// ---------------------------------------------------------------------------
__device__ __forceinline__ void tdm_load_2d_bf16(unsigned ldsAddr,
                                                 const bf16* gptr,
                                                 unsigned tileD0, unsigned tileD1,
                                                 unsigned tenD0, unsigned tenD1,
                                                 unsigned strideD0) {
  unsigned long long ga = (unsigned long long)(uintptr_t)gptr;
  v4u g0;
  g0[0] = 1u;                                            // count=1 (valid)
  g0[1] = ldsAddr;                                       // lds_addr [63:32]
  g0[2] = (unsigned)(ga & 0xffffffffu);                  // global_addr lo
  g0[3] = (unsigned)((ga >> 32) & 0x1ffffffu) | (2u << 30);  // addr hi | type=2
  v8i g1;
  g1[0] = (int)(1u << 16);                               // data_size=1 (2B)
  g1[1] = (int)((tenD0 & 0xffffu) << 16);                // tensor_dim0[15:0]
  g1[2] = (int)(((tenD0 >> 16) & 0xffffu) | ((tenD1 & 0xffffu) << 16));
  g1[3] = (int)(((tenD1 >> 16) & 0xffffu) | ((tileD0 & 0xffffu) << 16));
  g1[4] = (int)(tileD1 & 0xffffu);                       // tile_dim1, tile_dim2=0
  g1[5] = (int)strideD0;                                 // dim0_stride[31:0]
  g1[6] = 0;                                             // stride hi, dim1_stride lo
  g1[7] = 0;
  v4i g2 = {0, 0, 0, 0};
  v4i g3 = {0, 0, 0, 0};
  v8i g4 = {0, 0, 0, 0, 0, 0, 0, 0};
  __builtin_amdgcn_tensor_load_to_lds(g0, g1, g2, g3, g4, 0);
}

// ---------------------------------------------------------------------------
// fp32 -> bf16 pack (4 elements / thread)
// ---------------------------------------------------------------------------
__global__ __launch_bounds__(256) void k_cvt_bf16(const float* __restrict__ in,
                                                  bf16* __restrict__ out, int n) {
  int i = (blockIdx.x * 256 + threadIdx.x) * 4;
  if (i >= n) return;
  v4f x = *(const v4f*)(in + i);
  v4bf y;
#pragma unroll
  for (int j = 0; j < 4; ++j) y[j] = (bf16)x[j];
  *(v4bf*)(out + i) = y;
}

// ---------------------------------------------------------------------------
// Weight transpose+convert:  Wt[o][i] = (bf16) W[i][o]   (both 2048x2048)
// ---------------------------------------------------------------------------
__global__ __launch_bounds__(256) void k_wt(const float* __restrict__ W,
                                            bf16* __restrict__ Wt) {
  __shared__ float tile[32][33];
  const int bx = blockIdx.x * 32, by = blockIdx.y * 32;
  const int tx = threadIdx.x, ty = threadIdx.y;  // block (32, 8)
#pragma unroll
  for (int r = 0; r < 32; r += 8)
    tile[ty + r][tx] = W[(size_t)(by + ty + r) * DIMS + bx + tx];
  __syncthreads();
#pragma unroll
  for (int r = 0; r < 32; r += 8)
    Wt[(size_t)(bx + ty + r) * DIMS + by + tx] = (bf16)tile[tx][ty + r];
}

// ---------------------------------------------------------------------------
// GEMM:  out[M,N] = X[M,K(2048)] @ Wt[N,K]^T     (M=16384, N=2048)
// Block tile 128x128, BK=32, 8 waves (4x2), wave tile 32x64 (2x4 WMMA tiles).
// Tiles are DMAed into LDS by the Tensor Data Mover (wave 0 drives it),
// double buffered: TDM fills buffer (i+1)&1 while WMMAs consume buffer i&1.
// ---------------------------------------------------------------------------
template <bool OUT_F32>
__global__ __launch_bounds__(256) void k_gemm(const bf16* __restrict__ X,
                                              const bf16* __restrict__ Wt,
                                              void* __restrict__ outp) {
  __shared__ bf16 Xs[2][128][32];
  __shared__ bf16 Ws[2][128][32];

  const int rowBase = blockIdx.y * 128;
  const int colBase = blockIdx.x * 128;
  const int t    = threadIdx.x;
  const int lane = t & 31;
  const int wid  = t >> 5;
  const int wm   = wid >> 1;       // 0..3  (M direction)
  const int wn   = wid & 1;        // 0..1  (N direction)
  const int m    = lane & 15;
  const int hl   = lane >> 4;      // lane half

  v8f acc[2][4];
#pragma unroll
  for (int mt = 0; mt < 2; ++mt)
#pragma unroll
    for (int nt = 0; nt < 4; ++nt)
#pragma unroll
      for (int r = 0; r < 8; ++r) acc[mt][nt][r] = 0.0f;

  const bf16* xTile = X  + (size_t)rowBase * DIMS;  // + k0
  const bf16* wTile = Wt + (size_t)colBase * DIMS;  // + k0

  // Prologue: DMA first K-slice into buffer 0.
  if (wid == 0) {
    tdm_load_2d_bf16(lds_addr32(&Xs[0][0][0]), xTile, 32, 128, DIMS, NTOK, DIMS);
    tdm_load_2d_bf16(lds_addr32(&Ws[0][0][0]), wTile, 32, 128, DIMS, DIMS, DIMS);
    __builtin_amdgcn_s_wait_tensorcnt(0);
  }
  __syncthreads();

  int buf = 0;
  for (int k0 = 0; k0 < DIMS; k0 += 32, buf ^= 1) {
    const bool more = (k0 + 32) < DIMS;
    if (wid == 0 && more) {   // kick DMA for the next slice into buf^1
      tdm_load_2d_bf16(lds_addr32(&Xs[buf ^ 1][0][0]), xTile + k0 + 32,
                       32, 128, DIMS, NTOK, DIMS);
      tdm_load_2d_bf16(lds_addr32(&Ws[buf ^ 1][0][0]), wTile + k0 + 32,
                       32, 128, DIMS, DIMS, DIMS);
    }

    v16bf a[2], b[4];
#pragma unroll
    for (int mt = 0; mt < 2; ++mt) {
      const bf16* xr = &Xs[buf][wm * 32 + mt * 16 + m][0];
      a[mt] = join8(*(const v8bf*)(xr + hl * 16),
                    *(const v8bf*)(xr + hl * 16 + 8));
    }
#pragma unroll
    for (int nt = 0; nt < 4; ++nt) {
      const bf16* wr = &Ws[buf][wn * 64 + nt * 16 + m][0];
      b[nt] = join8(*(const v8bf*)(wr + hl * 16),
                    *(const v8bf*)(wr + hl * 16 + 8));
    }
#pragma unroll
    for (int mt = 0; mt < 2; ++mt)
#pragma unroll
      for (int nt = 0; nt < 4; ++nt)
        acc[mt][nt] = wmma_bf16(a[mt], b[nt], acc[mt][nt]);

    if (wid == 0 && more) __builtin_amdgcn_s_wait_tensorcnt(0);
    __syncthreads();
  }

  float* outF = (float*)outp;
  bf16*  outB = (bf16*)outp;
#pragma unroll
  for (int mt = 0; mt < 2; ++mt)
#pragma unroll
    for (int nt = 0; nt < 4; ++nt)
#pragma unroll
      for (int r = 0; r < 8; ++r) {
        int row = rowBase + wm * 32 + mt * 16 + r + hl * 8;
        int col = colBase + wn * 64 + nt * 16 + m;
        if (OUT_F32) outF[(size_t)row * DIMS + col] = acc[mt][nt][r];
        else         outB[(size_t)row * DIMS + col] = (bf16)acc[mt][nt][r];
      }
}

// ---------------------------------------------------------------------------
// In-place RoPE on a bf16 [NTOK, DIMS] buffer (one thread per (token,head,j))
// ---------------------------------------------------------------------------
__global__ __launch_bounds__(256) void k_rope(bf16* __restrict__ buf) {
  int idx = blockIdx.x * 256 + threadIdx.x;   // < NTOK * HEADS * 64
  int row = idx >> 10;                        // token row (b*S + s)
  int hj  = idx & 1023;
  int h   = hj >> 6;
  int j   = hj & 63;
  int s   = row & (SEQ - 1);
  float f = (float)s * __powf(10000.0f, -(float)j * (1.0f / 64.0f));
  float sn, cs;
  __sincosf(f, &sn, &cs);
  bf16* p = buf + (size_t)row * DIMS + h * HEAD_DIM + j;
  float x1 = (float)p[0], x2 = (float)p[64];
  p[0]  = (bf16)(x1 * cs - x2 * sn);
  p[64] = (bf16)(x2 * cs + x1 * sn);
}

// ---------------------------------------------------------------------------
// Attention: one block per (b, chunk, head). 512 threads = 16 waves.
// Wave w owns queries [w*16, w*16+16). Streams 16 key-blocks of 32 keys from
// the 512-key window (prev chunk + current chunk) with online softmax.
// ---------------------------------------------------------------------------
__global__ __launch_bounds__(512) void k_attn(const bf16* __restrict__ Qb,
                                              const bf16* __restrict__ Kb,
                                              const bf16* __restrict__ Vb,
                                              bf16* __restrict__ Ab) {
  __shared__ bf16 Ks[32][128];    // key-major K tile (== B-fragment layout)
  __shared__ bf16 Vt[128][32];    // dim-major V tile (== B-fragment layout)
  __shared__ bf16 Ps[16][512];    // per-wave P (16 rows x 32 keys)

  const int blk = blockIdx.x;
  const int h = blk & (HEADS - 1);
  const int c = (blk / HEADS) & (NCHUNK - 1);
  const int b = blk / (HEADS * NCHUNK);

  const int t    = threadIdx.x;
  const int lane = t & 31;
  const int w    = t >> 5;
  const int m    = lane & 15;
  const int hl   = lane >> 4;
  const int q0   = w * 16;

  // Preload Q fragments for all 4 k-steps (HEAD_DIM = 4 * 32).
  const bf16* qrow =
      Qb + (size_t)(b * SEQ + c * WIN + q0 + m) * DIMS + h * HEAD_DIM;
  v16bf qf[4];
#pragma unroll
  for (int ks = 0; ks < 4; ++ks)
    qf[ks] = join8(*(const v8bf*)(qrow + ks * 32 + hl * 16),
                   *(const v8bf*)(qrow + ks * 32 + hl * 16 + 8));

  v8f o[8];
#pragma unroll
  for (int nt = 0; nt < 8; ++nt)
#pragma unroll
    for (int r = 0; r < 8; ++r) o[nt][r] = 0.0f;
  float mrun[8], lrun[8];
#pragma unroll
  for (int r = 0; r < 8; ++r) { mrun[r] = -1.0e30f; lrun[r] = 0.0f; }

  const float scale = 0.08838834764831845f;  // 1/sqrt(HEAD_DIM)

  const int idx  = t * 8;        // cooperative loader: 512 thr * 8 elem = 4096
  const int lkey = idx >> 7;     // 0..31
  const int ldc  = idx & 127;    // 0..120 step 8

  for (int kb = 0; kb < 16; ++kb) {
    if (c == 0 && kb < 8) continue;  // chunk 0 has no previous chunk (uniform)
    __syncthreads();
    {
      int tok = (c - 1) * WIN + kb * 32 + lkey;   // >= 0 for every taken path
      size_t base = (size_t)(b * SEQ + tok) * DIMS + h * HEAD_DIM + ldc;
      v8bf kk = *(const v8bf*)(Kb + base);
      v8bf vv = *(const v8bf*)(Vb + base);
      *(v8bf*)&Ks[lkey][ldc] = kk;
#pragma unroll
      for (int i = 0; i < 8; ++i) Vt[ldc + i][lkey] = vv[i];
    }
    __syncthreads();

    // Wave-level skip when the whole key block is above this wave's diagonal.
    if (kb >= 8 && (kb * 32 - WIN) > (q0 + 15)) continue;

    // ---- scores: S = Q * K^T -------------------------------------------
    v8f s0, s1;
#pragma unroll
    for (int r = 0; r < 8; ++r) { s0[r] = 0.0f; s1[r] = 0.0f; }
#pragma unroll
    for (int ks = 0; ks < 4; ++ks) {
      v16bf b0 = join8(*(const v8bf*)&Ks[m][ks * 32 + hl * 16],
                       *(const v8bf*)&Ks[m][ks * 32 + hl * 16 + 8]);
      v16bf b1 = join8(*(const v8bf*)&Ks[m + 16][ks * 32 + hl * 16],
                       *(const v8bf*)&Ks[m + 16][ks * 32 + hl * 16 + 8]);
      s0 = wmma_bf16(qf[ks], b0, s0);
      s1 = wmma_bf16(qf[ks], b1, s1);
    }

    // ---- mask + online softmax ----------------------------------------
    float p0[8], p1[8];
#pragma unroll
    for (int r = 0; r < 8; ++r) {
      int qq = q0 + r + hl * 8;       // query index within chunk
      int j0 = kb * 32 + m;           // key index within 512-window
      int j1 = j0 + 16;
      bool ok0 = (j0 < WIN) ? (c > 0) : ((j0 - WIN) <= qq);
      bool ok1 = (j1 < WIN) ? (c > 0) : ((j1 - WIN) <= qq);
      float v0 = ok0 ? s0[r] * scale : -3.0e38f;
      float v1 = ok1 ? s1[r] * scale : -3.0e38f;
      float mx = fmaxf(v0, v1);
#pragma unroll
      for (int d = 8; d >= 1; d >>= 1) mx = fmaxf(mx, __shfl_xor(mx, d, 32));
      float mnew = fmaxf(mrun[r], mx);
      float corr = __expf(mrun[r] - mnew);
      float e0 = __expf(v0 - mnew);
      float e1 = __expf(v1 - mnew);
      float sum = e0 + e1;
#pragma unroll
      for (int d = 8; d >= 1; d >>= 1) sum += __shfl_xor(sum, d, 32);
      lrun[r] = lrun[r] * corr + sum;
      mrun[r] = mnew;
      p0[r] = e0;
      p1[r] = e1;
#pragma unroll
      for (int nt = 0; nt < 8; ++nt) o[nt][r] *= corr;
    }

    // ---- P through wave-private LDS (D-layout -> A-fragment) ----------
#pragma unroll
    for (int r = 0; r < 8; ++r) {
      Ps[w][(r + hl * 8) * 32 + m]      = (bf16)p0[r];
      Ps[w][(r + hl * 8) * 32 + m + 16] = (bf16)p1[r];
    }
    __asm__ volatile("" ::: "memory");  // keep LDS write->read ordered
    v16bf pa = join8(*(const v8bf*)&Ps[w][m * 32 + hl * 16],
                     *(const v8bf*)&Ps[w][m * 32 + hl * 16 + 8]);

    // ---- O += P * V ----------------------------------------------------
#pragma unroll
    for (int nt = 0; nt < 8; ++nt) {
      const bf16* vr = &Vt[nt * 16 + m][hl * 16];
      v16bf bv = join8(*(const v8bf*)vr, *(const v8bf*)(vr + 8));
      o[nt] = wmma_bf16(pa, bv, o[nt]);
    }
  }

  // ---- epilogue: normalize and store ----------------------------------
  float inv[8];
#pragma unroll
  for (int r = 0; r < 8; ++r) inv[r] = 1.0f / lrun[r];
  bf16* arow = Ab + (size_t)(b * SEQ + c * WIN + q0) * DIMS + h * HEAD_DIM;
#pragma unroll
  for (int nt = 0; nt < 8; ++nt)
#pragma unroll
    for (int r = 0; r < 8; ++r)
      arow[(size_t)(r + hl * 8) * DIMS + nt * 16 + m] =
          (bf16)(o[nt][r] * inv[r]);
}

// ---------------------------------------------------------------------------
// Host launcher
// ---------------------------------------------------------------------------
extern "C" void kernel_launch(void* const* d_in, const int* in_sizes, int n_in,
                              void* d_out, int out_size, void* d_ws,
                              size_t ws_size, hipStream_t stream) {
  const float* hidden = (const float*)d_in[0];
  const float* Wq = (const float*)d_in[1];
  const float* Wk = (const float*)d_in[2];
  const float* Wv = (const float*)d_in[3];
  const float* Wo = (const float*)d_in[4];

  char* ws = (char*)d_ws;
  size_t off = 0;
  const size_t actBytes = (size_t)NTOK * DIMS * sizeof(bf16);  // 64 MB
  const size_t wBytes   = (size_t)DIMS * DIMS * sizeof(bf16);  // 8 MB
  bf16* Xb  = (bf16*)(ws + off); off += actBytes;
  bf16* Wqt = (bf16*)(ws + off); off += wBytes;
  bf16* Wkt = (bf16*)(ws + off); off += wBytes;
  bf16* Wvt = (bf16*)(ws + off); off += wBytes;
  bf16* Wot = (bf16*)(ws + off); off += wBytes;
  bf16* Qb  = (bf16*)(ws + off); off += actBytes;
  bf16* Kb  = (bf16*)(ws + off); off += actBytes;
  bf16* Vb  = (bf16*)(ws + off); off += actBytes;
  bf16* Ab  = (bf16*)(ws + off); off += actBytes;

  // 1) pack hidden states to bf16
  int nElem = NTOK * DIMS;
  k_cvt_bf16<<<nElem / 1024, 256, 0, stream>>>(hidden, Xb, nElem);

  // 2) transpose + convert weights
  dim3 tb(32, 8), tg(DIMS / 32, DIMS / 32);
  k_wt<<<tg, tb, 0, stream>>>(Wq, Wqt);
  k_wt<<<tg, tb, 0, stream>>>(Wk, Wkt);
  k_wt<<<tg, tb, 0, stream>>>(Wv, Wvt);
  k_wt<<<tg, tb, 0, stream>>>(Wo, Wot);

  // 3) Q/K/V projections (bf16 out)
  dim3 gg(DIMS / 128, NTOK / 128);
  k_gemm<false><<<gg, 256, 0, stream>>>(Xb, Wqt, Qb);
  k_gemm<false><<<gg, 256, 0, stream>>>(Xb, Wkt, Kb);
  k_gemm<false><<<gg, 256, 0, stream>>>(Xb, Wvt, Vb);

  // 4) RoPE on Q and K (in place)
  int nRope = NTOK * HEADS * 64;
  k_rope<<<nRope / 256, 256, 0, stream>>>(Qb);
  k_rope<<<nRope / 256, 256, 0, stream>>>(Kb);

  // 5) block sliding-window attention
  k_attn<<<BATCH * NCHUNK * HEADS, 512, 0, stream>>>(Qb, Kb, Vb, Ab);

  // 6) output projection (fp32 out)
  k_gemm<true><<<gg, 256, 0, stream>>>(Ab, Wot, d_out);
}
